// TSQBEVCore_5299989643284
// MI455X (gfx1250) — matrix operand order repaired and checked by reading, not compiled, not run on hardware
//
#include <hip/hip_runtime.h>

typedef __bf16 bf16;
typedef __attribute__((ext_vector_type(16))) __bf16 v16bf;
typedef __attribute__((ext_vector_type(8)))  __bf16 v8bf;
typedef __attribute__((ext_vector_type(8)))  float  v8f;

#define CDIM 256
#define QN   900
#define BQ   1800
#define HDIM 32
#define MT   113   // ceil(1800/16) output-row tiles for GEMMs
#define QT   57    // ceil(900/16) query tiles
#define KCH  29    // ceil(900/32) 32-key chunks
#define VKP  928   // padded key stride for transposed V (KCH*32)

static __device__ __forceinline__ int imin(int a, int b) { return a < b ? a : b; }

// ---------------------------------------------------------------------------
// Transpose per-(b,v) feature map [C, HW] f32 -> [HW, C] f16 (coalesced gather
// layout for the sampler).  grid: (HW/32, C/32, B*V), block (32,8).
// ---------------------------------------------------------------------------
__global__ void k_transpose(const float* __restrict__ src, _Float16* __restrict__ dst, int HW) {
  __shared__ float tile[32][33];
  int bv = blockIdx.z;
  const float* s = src + (size_t)bv * CDIM * HW;
  _Float16*    d = dst + (size_t)bv * HW * CDIM;
  int hw0 = blockIdx.x * 32, c0 = blockIdx.y * 32;
  int tx = threadIdx.x, ty = threadIdx.y;
#pragma unroll
  for (int j = 0; j < 32; j += 8) {
    int c = c0 + ty + j, hw = hw0 + tx;
    if (hw < HW) tile[ty + j][tx] = s[(size_t)c * HW + hw];
  }
  __syncthreads();
#pragma unroll
  for (int j = 0; j < 32; j += 8) {
    int hw = hw0 + ty + j, c = c0 + tx;
    if (hw < HW) d[(size_t)hw * CDIM + c] = (_Float16)tile[tx][ty + j];
  }
}

// ---------------------------------------------------------------------------
// Sampler: one block per (b,q); 256 threads = channels.  Threads 0..23 compute
// the 24 (view,keypoint) projections into LDS (pixel coords are level-
// independent: x = u/z exactly), then all threads do bilinear gathers from the
// pixel-major f16 feature maps for both levels.
// ---------------------------------------------------------------------------
__global__ void k_sampler(const float* __restrict__ refs,
                          const float* __restrict__ intr,
                          const float* __restrict__ extr,
                          const _Float16* __restrict__ fT0,
                          const _Float16* __restrict__ fT1,
                          float* __restrict__ sampled,
                          bf16* __restrict__ ctx) {
  __shared__ float spx[24], spy[24], sval[24];
  int q = blockIdx.x, b = blockIdx.y, c = threadIdx.x;
  if (threadIdx.x < 24) {
    int v = threadIdx.x >> 2, kk = threadIdx.x & 3;
    const float kx[4] = {0.f, 2.f, 0.f, -2.f};
    const float ky[4] = {0.f, 0.f, 2.f, 0.f};
    const float* rp = refs + ((size_t)b * QN + q) * 3;
    float rx = rp[0] + kx[kk], ry = rp[1] + ky[kk], rz = rp[2];
    const float* E = extr + (size_t)(b * 6 + v) * 16;
    float cx = E[0] * rx + E[1] * ry + E[2]  * rz + E[3];
    float cy = E[4] * rx + E[5] * ry + E[6]  * rz + E[7];
    float cz = E[8] * rx + E[9] * ry + E[10] * rz + E[11];
    const float* Km = intr + (size_t)(b * 6 + v) * 9;
    float u  = Km[0] * cx + Km[1] * cy + Km[2] * cz;
    float vv = Km[3] * cx + Km[4] * cy + Km[5] * cz;
    float w  = Km[6] * cx + Km[7] * cy + Km[8] * cz;
    float zs = (fabsf(w) > 1e-6f) ? w : 1e-6f;
    spx[threadIdx.x] = u / zs;
    spy[threadIdx.x] = vv / zs;
    sval[threadIdx.x] = (w > 0.f) ? 1.f : 0.f;
  }
  __syncthreads();
  float cnt = 0.f;
#pragma unroll
  for (int s = 0; s < 24; ++s) cnt += sval[s];
  cnt = fmaxf(cnt, 1.f);
  float acc = 0.f;
#pragma unroll
  for (int lvl = 0; lvl < 2; ++lvl) {
    const int H = lvl ? 32 : 64, W = lvl ? 88 : 176;
    const _Float16* fT = lvl ? fT1 : fT0;
    for (int s = 0; s < 24; ++s) {
      if (sval[s] == 0.f) continue;
      float x = spx[s], y = spy[s];
      if (!(x > -1.f && x < (float)W && y > -1.f && y < (float)H)) continue;
      float x0f = floorf(x), y0f = floorf(y);
      int x0 = (int)x0f, y0 = (int)y0f;
      float wx1 = x - x0f, wy1 = y - y0f;
      const _Float16* base = fT + ((size_t)(b * 6 + (s >> 2)) * H * W) * CDIM + c;
      float c00 = 0.f, c10 = 0.f, c01 = 0.f, c11 = 0.f;
      bool xin0 = (x0 >= 0), xin1 = (x0 + 1 <= W - 1);
      bool yin0 = (y0 >= 0), yin1 = (y0 + 1 <= H - 1);
      if (xin0 && yin0) c00 = (float)base[((size_t)y0 * W + x0) * CDIM];
      if (xin1 && yin0) c10 = (float)base[((size_t)y0 * W + x0 + 1) * CDIM];
      if (xin0 && yin1) c01 = (float)base[((size_t)(y0 + 1) * W + x0) * CDIM];
      if (xin1 && yin1) c11 = (float)base[((size_t)(y0 + 1) * W + x0 + 1) * CDIM];
      acc += (c00 * (1.f - wx1) + c10 * wx1) * (1.f - wy1)
           + (c01 * (1.f - wx1) + c11 * wx1) * wy1;
    }
  }
  float outv = acc / cnt * 0.5f;
  size_t idx = ((size_t)b * QN + q) * CDIM + c;
  sampled[idx] = outv;
  ctx[idx] = (bf16)outv;
}

// ---------------------------------------------------------------------------
__global__ void k_cvt_bf16(const float* __restrict__ src, bf16* __restrict__ dst, int n) {
  int i = blockIdx.x * blockDim.x + threadIdx.x;
  if (i < n) dst[i] = (bf16)src[i];
}

// ---------------------------------------------------------------------------
// WMMA GEMM: out[m,n] = sum_k X[m,k] * W[n,k] + bias[n].
// One wave per 16x16 output tile; K = 256 in 8 steps of v_wmma_f32_16x16x32_bf16.
// A fragment (16x32) and B fragment (32x16) use one contiguous k order,
// k = 16*half + e; each lane does contiguous 16B loads.
// Output modes: outb  = bf16 row-major [M, 256]
//               outf  = f32  row-major [M, 256]
//               outt  = bf16 transposed-V layout [B][256][VKP] (key-major)
// ---------------------------------------------------------------------------
__global__ void k_gemm(const bf16* __restrict__ X, const bf16* __restrict__ Wm,
                       const float* __restrict__ bias,
                       bf16* __restrict__ outb, float* __restrict__ outf,
                       bf16* __restrict__ outt, int M) {
  int wave = threadIdx.x >> 5;
  int lane = threadIdx.x & 31;
  int m16 = lane & 15, half = lane >> 4;
  int mtile = blockIdx.x;
  int ntile = blockIdx.y * 8 + wave;
  int row = imin(mtile * 16 + m16, M - 1);
  int n = ntile * 16 + m16;
  v8f acc = {};
#pragma unroll
  for (int k0 = 0; k0 < CDIM; k0 += 32) {
    int ka = k0 + half * 16;
    v8bf alo = *(const v8bf*)(X + (size_t)row * CDIM + ka);
    v8bf ahi = *(const v8bf*)(X + (size_t)row * CDIM + ka + 8);
    v16bf a;
#pragma unroll
    for (int i = 0; i < 8; ++i) { a[i] = alo[i]; a[i + 8] = ahi[i]; }
    int kb = k0 + half * 16;
    v8bf b0 = *(const v8bf*)(Wm + (size_t)n * CDIM + kb);
    v8bf b1 = *(const v8bf*)(Wm + (size_t)n * CDIM + kb + 8);
    v16bf bb;
#pragma unroll
    for (int i = 0; i < 8; ++i) { bb[i] = b0[i]; bb[i + 8] = b1[i]; }
    acc = __builtin_amdgcn_wmma_f32_16x16x32_bf16(false, a, false, bb, (short)0, acc,
                                                  false, false);
  }
  float bi = bias[n];
#pragma unroll
  for (int r = 0; r < 8; ++r) {
    int grow = mtile * 16 + half * 8 + r;
    if (grow < M) {
      float val = acc[r] + bi;
      if (outb) {
        outb[(size_t)grow * CDIM + n] = (bf16)val;
      } else if (outf) {
        outf[(size_t)grow * CDIM + n] = val;
      } else {
        int bb2 = grow / QN;
        int key = grow - bb2 * QN;
        outt[((size_t)(bb2 * CDIM) + n) * VKP + key] = (bf16)val;
      }
    }
  }
}

// ---------------------------------------------------------------------------
// Flash attention: one wave per (b,h,16-query tile).  Head dim 32 = one WMMA
// K-step.  Streams 29 chunks of 32 keys: 2 score WMMAs -> online softmax (row
// stats partitioned exactly like the C/D VGPR layout: rows r + half*8,
// shfl_xor reductions within 16-lane halves) -> LDS transpose of P into the A
// fragment layout -> 2 PV WMMAs accumulating into f32.  V is pre-transposed
// [B][C][VKP] so each PV B-fragment is two aligned 16B loads per lane; pad
// keys 900..927 carry exactly-zero probabilities (finite poison * 0 = 0).
// ---------------------------------------------------------------------------
__global__ void k_attn(const bf16* __restrict__ qp, const bf16* __restrict__ kp,
                       const bf16* __restrict__ vT, bf16* __restrict__ outp) {
  __shared__ __align__(16) bf16 plds[8 * 16 * 32];
  int wave = threadIdx.x >> 5, lane = threadIdx.x & 31;
  int m16 = lane & 15, half = lane >> 4;
  int qtile = blockIdx.x * 8 + wave;
  if (qtile >= QT) return;
  int bh = blockIdx.y;
  int b = bh >> 3, h = bh & 7;
  const float scale = 0.17677669529663687f;  // 1/sqrt(32)
  bf16* myp = plds + wave * (16 * 32);

  int qrow = b * QN + imin(qtile * 16 + m16, QN - 1);
  const bf16* qbase = qp + (size_t)qrow * CDIM + h * HDIM + half * 16;
  v8bf qlo = *(const v8bf*)(qbase);
  v8bf qhi = *(const v8bf*)(qbase + 8);
  v16bf aq;
#pragma unroll
  for (int i = 0; i < 8; ++i) { aq[i] = qlo[i]; aq[i + 8] = qhi[i]; }

  // per-column base pointers into transposed V (column = head dim)
  const bf16* v0base = vT + ((size_t)(b * CDIM) + h * HDIM + m16) * VKP + half * 16;
  const bf16* v1base = v0base + (size_t)16 * VKP;

  v8f o0 = {}, o1 = {};
  float m8[8], l8[8];
#pragma unroll
  for (int r = 0; r < 8; ++r) { m8[r] = -3.0e38f; l8[r] = 0.f; }

  for (int ch = 0; ch < KCH; ++ch) {
    int base = ch * 32;
    int key0 = base + m16, key1 = key0 + 16;
    int kr0 = b * QN + imin(key0, QN - 1);
    int kr1 = b * QN + imin(key1, QN - 1);
    const bf16* kb0 = kp + (size_t)kr0 * CDIM + h * HDIM + half * 16;
    const bf16* kb1 = kp + (size_t)kr1 * CDIM + h * HDIM + half * 16;
    v8bf t0 = *(const v8bf*)kb0, t1 = *(const v8bf*)(kb0 + 8);
    v8bf t2 = *(const v8bf*)kb1, t3 = *(const v8bf*)(kb1 + 8);
    v16bf bk0, bk1;
#pragma unroll
    for (int i = 0; i < 8; ++i) {
      bk0[i] = t0[i]; bk0[i + 8] = t1[i];
      bk1[i] = t2[i]; bk1[i + 8] = t3[i];
    }
    v8f s0 = {}, s1 = {};
    s0 = __builtin_amdgcn_wmma_f32_16x16x32_bf16(false, aq, false, bk0, (short)0, s0,
                                                 false, false);
    s1 = __builtin_amdgcn_wmma_f32_16x16x32_bf16(false, aq, false, bk1, (short)0, s1,
                                                 false, false);
    bool msk0 = key0 >= QN, msk1 = key1 >= QN;
#pragma unroll
    for (int r = 0; r < 8; ++r) {
      s0[r] = msk0 ? -1.0e30f : s0[r] * scale;
      s1[r] = msk1 ? -1.0e30f : s1[r] * scale;
    }
#pragma unroll
    for (int r = 0; r < 8; ++r) {
      float cm = fmaxf(s0[r], s1[r]);
      cm = fmaxf(cm, __shfl_xor(cm, 1));
      cm = fmaxf(cm, __shfl_xor(cm, 2));
      cm = fmaxf(cm, __shfl_xor(cm, 4));
      cm = fmaxf(cm, __shfl_xor(cm, 8));
      float nm = fmaxf(m8[r], cm);
      float alpha = __expf(m8[r] - nm);
      float p0 = __expf(s0[r] - nm);
      float p1 = __expf(s1[r] - nm);
      float rs = p0 + p1;
      rs += __shfl_xor(rs, 1); rs += __shfl_xor(rs, 2);
      rs += __shfl_xor(rs, 4); rs += __shfl_xor(rs, 8);
      l8[r] = l8[r] * alpha + rs;
      m8[r] = nm;
      o0[r] *= alpha;
      o1[r] *= alpha;
      int prow = half * 8 + r;
      myp[prow * 32 + m16] = (bf16)p0;
      myp[prow * 32 + 16 + m16] = (bf16)p1;
    }
    asm volatile("s_wait_dscnt 0" ::: "memory");
    const bf16* pb = myp + m16 * 32 + half * 16;
    v8bf plo = *(const v8bf*)pb;
    v8bf phi = *(const v8bf*)(pb + 8);
    v16bf ap;
#pragma unroll
    for (int i = 0; i < 8; ++i) { ap[i] = plo[i]; ap[i + 8] = phi[i]; }
    // V B-fragments: element e <-> key base+half*16+e, contiguous in vT
    v8bf u0 = *(const v8bf*)(v0base + base);
    v8bf u1 = *(const v8bf*)(v0base + base + 8);
    v8bf u2 = *(const v8bf*)(v1base + base);
    v8bf u3 = *(const v8bf*)(v1base + base + 8);
    v16bf bv0, bv1;
#pragma unroll
    for (int i = 0; i < 8; ++i) {
      bv0[i] = u0[i]; bv0[i + 8] = u1[i];
      bv1[i] = u2[i]; bv1[i + 8] = u3[i];
    }
    o0 = __builtin_amdgcn_wmma_f32_16x16x32_bf16(false, ap, false, bv0, (short)0, o0,
                                                 false, false);
    o1 = __builtin_amdgcn_wmma_f32_16x16x32_bf16(false, ap, false, bv1, (short)0, o1,
                                                 false, false);
  }
#pragma unroll
  for (int r = 0; r < 8; ++r) {
    int grow = qtile * 16 + half * 8 + r;
    if (grow < QN) {
      float inv = 1.f / fmaxf(l8[r], 1e-20f);
      size_t o = (size_t)(b * QN + grow) * CDIM + h * HDIM;
      outp[o + m16] = (bf16)(o0[r] * inv);
      outp[o + 16 + m16] = (bf16)(o1[r] * inv);
    }
  }
}

// ---------------------------------------------------------------------------
extern "C" void kernel_launch(void* const* d_in, const int* in_sizes, int n_in,
                              void* d_out, int out_size, void* d_ws, size_t ws_size,
                              hipStream_t stream) {
  (void)in_sizes; (void)n_in; (void)out_size; (void)ws_size;
  const float* features0 = (const float*)d_in[0];
  const float* features1 = (const float*)d_in[1];
  const float* refs      = (const float*)d_in[2];
  const float* intr      = (const float*)d_in[3];
  const float* extr      = (const float*)d_in[4];
  const float* queries   = (const float*)d_in[5];
  const float* Wq = (const float*)d_in[6];  const float* bq = (const float*)d_in[7];
  const float* Wk = (const float*)d_in[8];  const float* bk = (const float*)d_in[9];
  const float* Wv = (const float*)d_in[10]; const float* bv = (const float*)d_in[11];
  const float* Wo = (const float*)d_in[12]; const float* bo = (const float*)d_in[13];
  float* out = (float*)d_out;

  char* ws = (char*)d_ws;
  size_t off = 0;
  auto take = [&](size_t bytes) {
    char* p = ws + off;
    off += (bytes + 255) & ~(size_t)255;
    return p;
  };
  _Float16* fT0 = (_Float16*)take((size_t)12 * 11264 * CDIM * 2);  // [B,V,H0,W0,C] f16
  _Float16* fT1 = (_Float16*)take((size_t)12 * 2816 * CDIM * 2);   // [B,V,H1,W1,C] f16
  float* sampled = (float*)take((size_t)BQ * CDIM * 4);
  bf16* ctx = (bf16*)take((size_t)BQ * CDIM * 2);
  bf16* qin = (bf16*)take((size_t)BQ * CDIM * 2);
  bf16* wqb = (bf16*)take((size_t)CDIM * CDIM * 2);
  bf16* wkb = (bf16*)take((size_t)CDIM * CDIM * 2);
  bf16* wvb = (bf16*)take((size_t)CDIM * CDIM * 2);
  bf16* wob = (bf16*)take((size_t)CDIM * CDIM * 2);
  bf16* qpj = (bf16*)take((size_t)BQ * CDIM * 2);
  bf16* kpj = (bf16*)take((size_t)BQ * CDIM * 2);
  bf16* vTt = (bf16*)take((size_t)2 * CDIM * VKP * 2);             // transposed V
  bf16* oat = (bf16*)take((size_t)BQ * CDIM * 2);

  // 1) channel-major -> pixel-major f16 feature transpose (coalesces the gather)
  k_transpose<<<dim3(11264 / 32, 8, 12), dim3(32, 8), 0, stream>>>(features0, fT0, 11264);
  k_transpose<<<dim3(2816 / 32, 8, 12), dim3(32, 8), 0, stream>>>(features1, fT1, 2816);
  // 2) projection + bilinear sampling -> sampled f32 and context bf16
  k_sampler<<<dim3(QN, 2), 256, 0, stream>>>(refs, intr, extr, fT0, fT1, sampled, ctx);
  // 3) bf16 conversions for WMMA operands
  int nq = BQ * CDIM;
  k_cvt_bf16<<<(nq + 255) / 256, 256, 0, stream>>>(queries, qin, nq);
  k_cvt_bf16<<<(65536 + 255) / 256, 256, 0, stream>>>(Wq, wqb, 65536);
  k_cvt_bf16<<<(65536 + 255) / 256, 256, 0, stream>>>(Wk, wkb, 65536);
  k_cvt_bf16<<<(65536 + 255) / 256, 256, 0, stream>>>(Wv, wvb, 65536);
  k_cvt_bf16<<<(65536 + 255) / 256, 256, 0, stream>>>(Wo, wob, 65536);
  // 4) Q/K/V projections (WMMA GEMM); V stored transposed for the PV B-fragment
  k_gemm<<<dim3(MT, 2), 256, 0, stream>>>(qin, wqb, bq, qpj, nullptr, nullptr, BQ);
  k_gemm<<<dim3(MT, 2), 256, 0, stream>>>(ctx, wkb, bk, kpj, nullptr, nullptr, BQ);
  k_gemm<<<dim3(MT, 2), 256, 0, stream>>>(ctx, wvb, bv, nullptr, nullptr, vTt, BQ);
  // 5) flash attention (WMMA QK^T + online softmax + WMMA PV)
  k_attn<<<dim3((QT + 7) / 8, 16), 256, 0, stream>>>(qpj, kpj, vTt, oat);
  // 6) output projection (WMMA GEMM, f32 out -> d_out)
  k_gemm<<<dim3(MT, 2), 256, 0, stream>>>(oat, wob, bo, nullptr, out, nullptr, BQ);
}
